// RACAttention_34806414967150
// MI455X (gfx1250) — matrix operand order repaired and checked by reading, not compiled, not run on hardware
//
#include <hip/hip_runtime.h>
#include <hip/hip_bf16.h>

// ---------------------------------------------------------------------------
// Causal MHA forward for B=2, T=2048, D=1024, H=16, DH=64 on gfx1250 (CDNA5).
// Pipeline: f32->f16 convert, QKV GEMM (WMMA + TDM double-buffered LDS),
// flash attention (WMMA, K-fragment software pipeline), output GEMM.
// ---------------------------------------------------------------------------

typedef _Float16 v16h __attribute__((ext_vector_type(16)));
typedef _Float16 h8   __attribute__((ext_vector_type(8)));
typedef _Float16 h4   __attribute__((ext_vector_type(4)));
typedef float    v8f  __attribute__((ext_vector_type(8)));
typedef unsigned int u32x4 __attribute__((ext_vector_type(4)));
typedef int      i32x4 __attribute__((ext_vector_type(4)));
typedef int      i32x8 __attribute__((ext_vector_type(8)));

union Frag16 { v16h v; h8 h[2]; };

static __device__ __forceinline__ v8f wmma16(v16h a, v16h b, v8f c) {
  // D = A(16x32 f16) * B(32x16 f16) + C(16x16 f32)
  return __builtin_amdgcn_wmma_f32_16x16x32_f16(
      /*neg_a=*/false, a, /*neg_b=*/false, b,
      /*c_mod=*/(short)0, c, /*reuse_a=*/false, /*reuse_b=*/false);
}

// ---------------------------------------------------------------------------
// Tensor Data Mover helpers (gfx1250). 2D tile load: tile_dim0 x tile_dim1
// elements of data_size=2B from a row-major tensor with stride tensor_dim0.
// ---------------------------------------------------------------------------
#if __has_builtin(__builtin_amdgcn_tensor_load_to_lds) && \
    __has_builtin(__builtin_amdgcn_s_wait_tensorcnt)
#define HAVE_TDM 1
#else
#define HAVE_TDM 0
#endif

#if HAVE_TDM
static __device__ __forceinline__ u32x4 tdm_g0(unsigned lds_off, const void* gaddr) {
  unsigned long long ga = (unsigned long long)gaddr;
  u32x4 g0;
  g0.x = 1u;                                               // count=1 (valid D#)
  g0.y = lds_off;                                          // lds_addr [63:32]
  g0.z = (unsigned)(ga & 0xffffffffu);                     // global_addr lo
  g0.w = (unsigned)((ga >> 32) & 0x01ffffffu) | (2u << 30);// addr hi | type=2
  return g0;
}

static __device__ __forceinline__ i32x8 tdm_g1(unsigned dim0, unsigned dim1,
                                               unsigned tile0, unsigned tile1,
                                               unsigned stride0) {
  i32x8 g1;
  g1[0] = 0x10000;                                         // data_size=1 -> 2B
  g1[1] = (int)((dim0 & 0xffffu) << 16);                   // tensor_dim0 lo16
  g1[2] = (int)((dim0 >> 16) | ((dim1 & 0xffffu) << 16));  // dim0 hi / dim1 lo
  g1[3] = (int)((dim1 >> 16) | (tile0 << 16));             // dim1 hi / tile0
  g1[4] = (int)(tile1 & 0xffffu);                          // tile1, tile2=0
  g1[5] = (int)stride0;                                    // dim0 stride lo32
  g1[6] = 0;
  g1[7] = 0;
  return g1;
}

static __device__ __forceinline__ void tdm_issue(u32x4 g0, i32x8 g1) {
  i32x4 z4 = {0, 0, 0, 0};
#if defined(__clang_major__) && (__clang_major__ >= 23)
  i32x8 z8 = {0, 0, 0, 0, 0, 0, 0, 0};
  __builtin_amdgcn_tensor_load_to_lds(g0, g1, z4, z4, z8, 0);
#else
  __builtin_amdgcn_tensor_load_to_lds(g0, g1, z4, z4, 0);
#endif
}
#endif  // HAVE_TDM

// ---------------------------------------------------------------------------
// Kernel 0: f32 -> f16 convert (vectorized x4)
// ---------------------------------------------------------------------------
__global__ void cvt_f32_f16(const float* __restrict__ in,
                            _Float16* __restrict__ out, int n) {
  int i = (blockIdx.x * blockDim.x + threadIdx.x) * 4;
  if (i + 3 < n) {
    float4 v = *(const float4*)(in + i);
    h4 o = {(_Float16)v.x, (_Float16)v.y, (_Float16)v.z, (_Float16)v.w};
    *(h4*)(out + i) = o;
  }
}

// ---------------------------------------------------------------------------
// Kernel 1/3: C[M,N] = A[M,K] * W[N,K]^T + bias[N]
// Block = 256 threads = 8 waves; block tile 128x128; K-step 32.
// LDS tiles filled by the Tensor Data Mover (double-buffered: TDM for step
// i+1 is in flight while step i computes). Wave (wm,wn) owns 64x32 = 4x2
// WMMA tiles. OUT16 ? write f16 : write f32.
// ---------------------------------------------------------------------------
template <bool OUT16>
__global__ void __launch_bounds__(256, 1)
gemm_nt(const _Float16* __restrict__ A,
        const _Float16* __restrict__ W,
        const float* __restrict__ bias,
        void* __restrict__ Cout, int M, int N, int K) {
  __shared__ _Float16 As[2][128 * 32];
  __shared__ _Float16 Bs[2][128 * 32];
  const int tid = threadIdx.x;
  const int lane = tid & 31, lo = lane & 15, hi = lane >> 4;
  const int w = tid >> 5, wm = w >> 2, wn = w & 3;
  const int bm = blockIdx.y * 128, bn = blockIdx.x * 128;
  const int nk = K >> 5;

  v8f acc[4][2] = {};

#if HAVE_TDM
  const i32x8 g1A = tdm_g1((unsigned)K, (unsigned)M, 32u, 128u, (unsigned)K);
  const i32x8 g1B = tdm_g1((unsigned)K, (unsigned)N, 32u, 128u, (unsigned)K);
  auto issue = [&](int i) {
    const unsigned ldsA = (unsigned)(unsigned long long)(const void*)&As[i & 1][0];
    const unsigned ldsB = (unsigned)(unsigned long long)(const void*)&Bs[i & 1][0];
    tdm_issue(tdm_g0(ldsA, A + (size_t)bm * K + i * 32), g1A);
    tdm_issue(tdm_g0(ldsB, W + (size_t)bn * K + i * 32), g1B);
  };
  if (tid < 32) issue(0);
#endif

  for (int i = 0; i < nk; ++i) {
#if HAVE_TDM
    if (tid < 32) __builtin_amdgcn_s_wait_tensorcnt(0);
    __syncthreads();                       // tile i visible to all waves
    if (tid < 32 && i + 1 < nk) issue(i + 1);
#else
    // Fallback: per-thread vector fill, single-buffer semantics.
    __syncthreads();
    {
      const int lr = tid >> 1, lc = (tid & 1) * 16;
      const _Float16* sa = A + (size_t)(bm + lr) * K + i * 32 + lc;
      *(h8*)(&As[i & 1][lr * 32 + lc])     = *(const h8*)(sa);
      *(h8*)(&As[i & 1][lr * 32 + lc + 8]) = *(const h8*)(sa + 8);
      const _Float16* sb = W + (size_t)(bn + lr) * K + i * 32 + lc;
      *(h8*)(&Bs[i & 1][lr * 32 + lc])     = *(const h8*)(sb);
      *(h8*)(&Bs[i & 1][lr * 32 + lc + 8]) = *(const h8*)(sb + 8);
      if (i + 1 < nk) {
        __builtin_prefetch(sa + 32, 0, 0);
        __builtin_prefetch(sb + 32, 0, 0);
      }
    }
    __syncthreads();
#endif
    const _Float16* as = &As[i & 1][0];
    const _Float16* bs = &Bs[i & 1][0];

    Frag16 af[4], bf[2];
#pragma unroll
    for (int mt = 0; mt < 4; ++mt) {
      // A-frag (16x32 f16): lane row = lo, K halves hi*8 + {0..7, 16..23}
      const _Float16* p = as + (wm * 64 + mt * 16 + lo) * 32 + hi * 8;
      af[mt].h[0] = *(const h8*)(p);
      af[mt].h[1] = *(const h8*)(p + 16);
    }
#pragma unroll
    for (int nt = 0; nt < 2; ++nt) {
      // B-frag (32x16 f16): lane col = lo, K halves hi*8 + {0..7, 16..23}
      const _Float16* p = bs + (wn * 32 + nt * 16 + lo) * 32 + hi * 8;
      bf[nt].h[0] = *(const h8*)(p);
      bf[nt].h[1] = *(const h8*)(p + 16);
    }
#pragma unroll
    for (int mt = 0; mt < 4; ++mt) {
#pragma unroll
      for (int nt = 0; nt < 2; ++nt)
        acc[mt][nt] = wmma16(af[mt].v, bf[nt].v, acc[mt][nt]);
    }
  }

  // Epilogue: C layout -> lane holds rows r + 8*hi, col lo, per VGPR r.
#pragma unroll
  for (int nt = 0; nt < 2; ++nt) {
    const int n = bn + wn * 32 + nt * 16 + lo;
    const float bv = bias[n];
#pragma unroll
    for (int mt = 0; mt < 4; ++mt) {
#pragma unroll
      for (int r = 0; r < 8; ++r) {
        const int mrow = bm + wm * 64 + mt * 16 + r + 8 * hi;
        const float v = acc[mt][nt][r] + bv;
        if (OUT16)
          ((_Float16*)Cout)[(size_t)mrow * N + n] = (_Float16)v;
        else
          ((float*)Cout)[(size_t)mrow * N + n] = v;
      }
    }
  }
}

// ---------------------------------------------------------------------------
// Kernel 2: causal flash attention.
// qkv: [B, T, 3D] f16 (q | k | v per token). out: [B, T, D] f16.
// Grid: (T/128, H, B), 256 threads = 8 independent waves; each wave owns
// 16 query rows and streams 32-key chunks with online softmax.
// Only the K fragments are double-buffered in registers (V rows are issued
// at the top of process() and consumed after the softmax, so their latency
// is hidden without a second register set).
// ---------------------------------------------------------------------------
__global__ void __launch_bounds__(256, 1)
attn_kernel(const _Float16* __restrict__ qkv, _Float16* __restrict__ out) {
  constexpr int T = 2048, D = 1024, TOK = 3 * 1024;
  __shared__ _Float16 Plds[8][16 * 32];   // per-wave P (16q x 32k)
  __shared__ _Float16 Vlds[8][64 * 32];   // per-wave V^T (64dh x 32k)

  const int tid = threadIdx.x;
  const int w = tid >> 5, lane = tid & 31, lo = lane & 15, hi = lane >> 4;
  const int b = blockIdx.z, h = blockIdx.y;
  const int qbase = blockIdx.x * 128 + w * 16;

  const size_t bBase = (size_t)b * T * TOK;
  const _Float16* Qp = qkv + bBase + (size_t)(qbase + lo) * TOK + h * 64;
  const _Float16* Kbase = qkv + bBase + 1024 + h * 64;
  const _Float16* Vbase = qkv + bBase + 2048 + h * 64;
  _Float16* Pw = &Plds[w][0];
  _Float16* Vw = &Vlds[w][0];

  // Q A-fragments for the two DH=32 chunks (row = lo; K halves hi*8 pattern)
  Frag16 qf[2];
#pragma unroll
  for (int dc = 0; dc < 2; ++dc) {
    qf[dc].h[0] = *(const h8*)(Qp + dc * 32 + hi * 8);
    qf[dc].h[1] = *(const h8*)(Qp + dc * 32 + 16 + hi * 8);
  }

  v8f O[4] = {};
  float m[8], lsum[8];
#pragma unroll
  for (int r = 0; r < 8; ++r) { m[r] = -1e30f; lsum[r] = 0.0f; }

  auto loadK = [&](int k0, Frag16 (&kf)[2][2]) {
#pragma unroll
    for (int nt = 0; nt < 2; ++nt) {
      const _Float16* kp = Kbase + (size_t)(k0 + nt * 16 + lo) * TOK;
#pragma unroll
      for (int dc = 0; dc < 2; ++dc) {
        kf[nt][dc].h[0] = *(const h8*)(kp + dc * 32 + hi * 8);
        kf[nt][dc].h[1] = *(const h8*)(kp + dc * 32 + hi * 8 + 16);
      }
    }
  };

  auto process = [&](int k0, Frag16 (&kf)[2][2]) {
    // ---- issue V row loads first; consumed only after softmax ----
    h8 vr[8];
    {
      const _Float16* vp = Vbase + (size_t)(k0 + lane) * TOK;
#pragma unroll
      for (int i = 0; i < 8; ++i) vr[i] = *(const h8*)(vp + i * 8);
    }

    // ---- S = Q * K^T for two 16-key tiles (4 WMMAs) ----
    v8f S[2];
#pragma unroll
    for (int nt = 0; nt < 2; ++nt) {
      v8f s = {};
#pragma unroll
      for (int dc = 0; dc < 2; ++dc) s = wmma16(qf[dc].v, kf[nt][dc].v, s);
      S[nt] = s;
    }

    // ---- online softmax (rows live across 16-lane half-waves) ----
    float p0[8], p1[8];
#pragma unroll
    for (int r = 0; r < 8; ++r) {
      const int row = qbase + r + 8 * hi;
      float s0 = S[0][r] * 0.125f;            // SCALE = DH^-0.5
      float s1 = S[1][r] * 0.125f;
      if (k0 + lo > row)      s0 = -1e30f;    // causal mask
      if (k0 + 16 + lo > row) s1 = -1e30f;
      float mx = fmaxf(s0, s1);
#pragma unroll
      for (int d = 1; d < 16; d <<= 1) mx = fmaxf(mx, __shfl_xor(mx, d, 32));
      const float mn = fmaxf(m[r], mx);
      const float corr = __expf(m[r] - mn);
      m[r] = mn;
      lsum[r] *= corr;
#pragma unroll
      for (int t = 0; t < 4; ++t) O[t][r] *= corr;
      const float e0 = __expf(s0 - mn);
      const float e1 = __expf(s1 - mn);
      float ps = e0 + e1;
#pragma unroll
      for (int d = 1; d < 16; d <<= 1) ps += __shfl_xor(ps, d, 32);
      lsum[r] += ps;
      p0[r] = e0;
      p1[r] = e1;
    }

    // ---- write P (C-layout regs -> [q][k] LDS) ----
#pragma unroll
    for (int r = 0; r < 8; ++r) {
      const int q = r + 8 * hi;
      Pw[q * 32 + lo]      = (_Float16)p0[r];
      Pw[q * 32 + 16 + lo] = (_Float16)p1[r];
    }
    // ---- scatter V transposed: Vw[dh][key] ----
#pragma unroll
    for (int i = 0; i < 8; ++i) {
#pragma unroll
      for (int e = 0; e < 8; ++e) Vw[(i * 8 + e) * 32 + lane] = vr[i][e];
    }
    __asm__ volatile("" ::: "memory");  // same-wave LDS is in-order; keep RAW order

    // ---- O += P * V (4 WMMAs over DH tiles) ----
    Frag16 pf;
    pf.h[0] = *(const h8*)(Pw + lo * 32 + hi * 8);
    pf.h[1] = *(const h8*)(Pw + lo * 32 + 16 + hi * 8);
#pragma unroll
    for (int t = 0; t < 4; ++t) {
      Frag16 vf;
      vf.h[0] = *(const h8*)(Vw + (t * 16 + lo) * 32 + hi * 8);
      vf.h[1] = *(const h8*)(Vw + (t * 16 + lo) * 32 + hi * 8 + 16);
      O[t] = wmma16(pf.v, vf.v, O[t]);
    }
  };

  // ---- two-stage K-fragment pipeline over 32-key chunks (causal) ----
  const int nchunks = ((qbase + 15) >> 5) + 1;
  Frag16 kfA[2][2], kfB[2][2];
  loadK(0, kfA);
  int kc = 0;
  while (true) {
    if (kc + 1 < nchunks) loadK((kc + 1) * 32, kfB);
    process(kc * 32, kfA);
    if (++kc >= nchunks) break;
    if (kc + 1 < nchunks) loadK((kc + 1) * 32, kfA);
    process(kc * 32, kfB);
    if (++kc >= nchunks) break;
  }

  // ---- epilogue: O /= l, write f16 [B,T,D] ----
#pragma unroll
  for (int r = 0; r < 8; ++r) lsum[r] = 1.0f / lsum[r];
  _Float16* op = out + ((size_t)b * T + qbase) * D + h * 64;
#pragma unroll
  for (int t = 0; t < 4; ++t) {
#pragma unroll
    for (int r = 0; r < 8; ++r)
      op[(size_t)(r + 8 * hi) * D + t * 16 + lo] = (_Float16)(O[t][r] * lsum[r]);
  }
}

// ---------------------------------------------------------------------------
// Host-side launch
// ---------------------------------------------------------------------------
extern "C" void kernel_launch(void* const* d_in, const int* in_sizes, int n_in,
                              void* d_out, int out_size, void* d_ws, size_t ws_size,
                              hipStream_t stream) {
  (void)in_sizes; (void)n_in; (void)out_size; (void)ws_size;
  constexpr int B = 2, T = 2048, D = 1024;
  constexpr int M = B * T;          // 4096
  constexpr int N_QKV = 3 * D;      // 3072
  constexpr int K = D;              // 1024

  const float* x     = (const float*)d_in[0];
  const float* w_qkv = (const float*)d_in[1];
  const float* b_qkv = (const float*)d_in[2];
  const float* w_out = (const float*)d_in[3];
  const float* b_out = (const float*)d_in[4];

  size_t off = 0;
  auto carve = [&](size_t bytes) {
    void* p = (char*)d_ws + off;
    off += (bytes + 255) & ~(size_t)255;
    return p;
  };
  _Float16* xh    = (_Float16*)carve((size_t)M * D * 2);          // 8 MB
  _Float16* wqkvh = (_Float16*)carve((size_t)N_QKV * D * 2);      // 6 MB
  _Float16* wouth = (_Float16*)carve((size_t)D * D * 2);          // 2 MB
  _Float16* qkvh  = (_Float16*)carve((size_t)M * N_QKV * 2);      // 24 MB
  _Float16* attnh = (_Float16*)carve((size_t)M * D * 2);          // 8 MB

  // 0) downconvert inputs to f16
  {
    int n;
    n = M * D;
    cvt_f32_f16<<<(n / 4 + 255) / 256, 256, 0, stream>>>(x, xh, n);
    n = N_QKV * D;
    cvt_f32_f16<<<(n / 4 + 255) / 256, 256, 0, stream>>>(w_qkv, wqkvh, n);
    n = D * D;
    cvt_f32_f16<<<(n / 4 + 255) / 256, 256, 0, stream>>>(w_out, wouth, n);
  }

  // 1) QKV projection: [4096,1024] x [3072,1024]^T + b  -> f16
  gemm_nt<true><<<dim3(N_QKV / 128, M / 128), 256, 0, stream>>>(
      xh, wqkvh, b_qkv, (void*)qkvh, M, N_QKV, K);

  // 2) causal flash attention
  attn_kernel<<<dim3(T / 128, 16 /*H*/, B), 256, 0, stream>>>(qkvh, attnh);

  // 3) output projection: [4096,1024] x [1024,1024]^T + b -> f32 d_out
  gemm_nt<false><<<dim3(D / 128, M / 128), 256, 0, stream>>>(
      attnh, wouth, b_out, d_out, M, D, K);
}
